// MultiFisonAttention_73942156968307
// MI455X (gfx1250) — matrix operand order repaired and checked by reading, not compiled, not run on hardware
//
#include <hip/hip_runtime.h>
#include <hip/hip_bf16.h>

// ---------------------------------------------------------------------------
// MultiFisonAttention for MI455X (gfx1250, wave32, WMMA).
//
// Precision strategy: f32 inputs are converted to f16 at fragment-load time and
// accumulated in f32 via v_wmma_f32_16x16x32_f16 (the highest-K 16-bit WMMA).
// This keeps the big projection GEMMs near the matrix-op roofline instead of
// the 8x-slower f32 16x16x4 WMMA path, while f32 accumulation over K=768
// bounds the error.
//
// Fragment layouts per CDNA5 ISA 7.12.2 (wave32):
//   A (16x32 f16):  lane l -> row M = l%16; half = l/16.
//                   vector elems [0..7]  = K: k0 + 8*half + 0..7
//                   vector elems [8..15] = K: k0 + 16 + 8*half + 0..7
//   B (32x16 f16):  lane l -> col N = l%16; half = l/16.
//                   vector elems [0..7] = K: k0 + 8*half + 0..7, [8..15] = K: k0 + 16 + 8*half + 0..7
//   C/D (16x16 f32): vgpr v, lane l: M = v + 8*(l/16), N = l%16
// ---------------------------------------------------------------------------

typedef _Float16 f16;
typedef __attribute__((ext_vector_type(16))) _Float16 v16h;
typedef __attribute__((ext_vector_type(8)))  float    v8f;

#define D_MODEL 768
#define SEQ     18
#define BATCH   4096
#define M_ROWS  (BATCH * SEQ)          // 73728 (multiple of 128)
#define KSTEPS  (D_MODEL / 32)         // 24
#define SCALE_F 0.03608439182435161f   // 1/sqrt(768)
#define LN_EPS  1e-5f

// workspace layout (bytes)
#define WS_WQ   ((size_t)0)
#define WS_WK   ((size_t)(WS_WQ  + (size_t)D_MODEL * D_MODEL * 2))
#define WS_WV   ((size_t)(WS_WK  + (size_t)D_MODEL * D_MODEL * 2))
#define WS_WFC  ((size_t)(WS_WV  + (size_t)D_MODEL * D_MODEL * 2))
#define WS_QH   ((size_t)(WS_WFC + (size_t)D_MODEL * D_MODEL * 2))
#define WS_KH   ((size_t)(WS_QH  + (size_t)M_ROWS * D_MODEL * 2))
#define WS_VH   ((size_t)(WS_KH  + (size_t)M_ROWS * D_MODEL * 2))
#define WS_CTX  ((size_t)(WS_VH  + (size_t)M_ROWS * D_MODEL * 2))

#define OUT_ATTN_OFF ((size_t)BATCH * D_MODEL)   // attn_w follows `out` in d_out

union HVec {
  v16h  h;
  uint4 u[2];
};

static __device__ __forceinline__ v16h wmma_acc_dummy() { return v16h{}; }

// ---------------------------------------------------------------------------
// K0: convert the four weight matrices to f16 (read once per dispatch instead
// of streaming f32 weights through every GEMM inner loop).
// ---------------------------------------------------------------------------
__global__ __launch_bounds__(256) void cvt_weights(const float* __restrict__ wq,
                                                   const float* __restrict__ wk,
                                                   const float* __restrict__ wv,
                                                   const float* __restrict__ wfc,
                                                   f16* __restrict__ out) {
  const int n = D_MODEL * D_MODEL;
  int i = blockIdx.x * blockDim.x + threadIdx.x;
  if (i >= 4 * n) return;
  int which = i / n;
  int off   = i - which * n;
  const float* src = (which == 0) ? wq : (which == 1) ? wk : (which == 2) ? wv : wfc;
  out[i] = (f16)src[off];
}

// ---------------------------------------------------------------------------
// K1: P = cvt16(X) @ cvt16(W)^T, P stored f16.  M=73728, N=768, K=768.
// WG = 256 threads (8 waves), tile 128M x 64N; wave w owns rows
// [row0+16w, +16) and 4 accumulators covering the 64 N columns.
// ---------------------------------------------------------------------------
__global__ __launch_bounds__(256) void proj_gemm(const float* __restrict__ X,
                                                 const f16*  __restrict__ Wh,
                                                 f16*        __restrict__ Ph) {
  const int lane = threadIdx.x & 31;
  const int w    = threadIdx.x >> 5;
  const int half = lane >> 4;
  const int l16  = lane & 15;
  const int row0 = blockIdx.y * 128 + w * 16;
  const int col0 = blockIdx.x * 64;

  v8f acc[4];
#pragma unroll
  for (int j = 0; j < 4; ++j) acc[j] = v8f{};

  const float* aptr = X + (size_t)(row0 + l16) * D_MODEL + 8 * half;

  for (int kk = 0; kk < KSTEPS; ++kk) {
    const int k0 = kk * 32;
    // ---- A fragment: load 16 f32, convert to f16 inline ----
    float fbuf[16];
    *(float4*)(&fbuf[0])  = *(const float4*)(aptr + k0);
    *(float4*)(&fbuf[4])  = *(const float4*)(aptr + k0 + 4);
    *(float4*)(&fbuf[8])  = *(const float4*)(aptr + k0 + 16);
    *(float4*)(&fbuf[12]) = *(const float4*)(aptr + k0 + 20);
    v16h a;
#pragma unroll
    for (int i = 0; i < 16; ++i) a[i] = (f16)fbuf[i];

    // ---- 4 B fragments (two 16B chunks per lane, 32B apart, from weight row) ----
#pragma unroll
    for (int j = 0; j < 4; ++j) {
      HVec b;
      const f16* bp = Wh + (size_t)(col0 + j * 16 + l16) * D_MODEL + k0 + 8 * half;
      b.u[0] = *(const uint4*)(bp);
      b.u[1] = *(const uint4*)(bp + 16);
      acc[j] = __builtin_amdgcn_wmma_f32_16x16x32_f16(
          false, a, false, b.h, (short)0, acc[j], false, false);
    }
  }

  // store f16
#pragma unroll
  for (int j = 0; j < 4; ++j) {
#pragma unroll
    for (int v = 0; v < 8; ++v) {
      int r = row0 + v + 8 * half;
      int c = col0 + j * 16 + l16;
      Ph[(size_t)r * D_MODEL + c] = (f16)acc[j][v];
    }
  }
}

// ---------------------------------------------------------------------------
// K2: per-batch attention. 4 waves, each computing one 16x16 score tile of the
// 32x32 (18-padded) score matrix via WMMA over K=768. Invalid rows/cols are
// zeroed with bitmasks (no EXEC divergence around WMMA). Then softmax,
// attn_w = attn^T @ W_attn^T, context = attn_w^T @ V (VALU), ctx stored f16.
// ---------------------------------------------------------------------------
__global__ __launch_bounds__(128) void attention_kernel(
    const f16* __restrict__ Qh, const f16* __restrict__ Kh,
    const f16* __restrict__ Vh, const unsigned char* __restrict__ amask,
    const float* __restrict__ Wattn, f16* __restrict__ ctxh,
    float* __restrict__ out) {
  __shared__ float sc[32 * 32];
  __shared__ float aw[SEQ];

  const int b    = blockIdx.x;
  const int tid  = threadIdx.x;
  const int lane = tid & 31;
  const int wv   = tid >> 5;        // 0..3
  const int mt   = wv >> 1;         // query tile
  const int nt   = wv & 1;          // key tile
  const int half = lane >> 4;
  const int l16  = lane & 15;

  const int qrl = mt * 16 + l16;
  const int krl = nt * 16 + l16;
  const unsigned am = (qrl < SEQ) ? 0xFFFFFFFFu : 0u;
  const unsigned bm = (krl < SEQ) ? 0xFFFFFFFFu : 0u;
  const size_t qr = (size_t)(b * SEQ + (qrl < SEQ ? qrl : SEQ - 1)) * D_MODEL;
  const size_t kr = (size_t)(b * SEQ + (krl < SEQ ? krl : SEQ - 1)) * D_MODEL;

  v8f acc = v8f{};
  for (int kk = 0; kk < KSTEPS; ++kk) {
    const int k0 = kk * 32;
    HVec a, bb;
    const f16* ap = Qh + qr + k0 + 8 * half;
    a.u[0] = *(const uint4*)(ap);
    a.u[1] = *(const uint4*)(ap + 16);
    a.u[0].x &= am; a.u[0].y &= am; a.u[0].z &= am; a.u[0].w &= am;
    a.u[1].x &= am; a.u[1].y &= am; a.u[1].z &= am; a.u[1].w &= am;
    const f16* bp = Kh + kr + k0 + 8 * half;
    bb.u[0] = *(const uint4*)(bp);
    bb.u[1] = *(const uint4*)(bp + 16);
    bb.u[0].x &= bm; bb.u[0].y &= bm; bb.u[0].z &= bm; bb.u[0].w &= bm;
    bb.u[1].x &= bm; bb.u[1].y &= bm; bb.u[1].z &= bm; bb.u[1].w &= bm;
    acc = __builtin_amdgcn_wmma_f32_16x16x32_f16(
        false, a.h, false, bb.h, (short)0, acc, false, false);
  }
#pragma unroll
  for (int v = 0; v < 8; ++v) {
    int q = mt * 16 + v + 8 * half;
    int k = nt * 16 + l16;
    sc[q * 32 + k] = acc[v];
  }
  __syncthreads();

  // softmax over k<18, per query row (threads 0..17)
  if (tid < SEQ) {
    const int q = tid;
    float mx = -3.4e38f;
    for (int k = 0; k < SEQ; ++k) {
      float s = sc[q * 32 + k] * SCALE_F;
      if (amask[(size_t)b * SEQ * SEQ + q * SEQ + k]) s = -1e9f;
      sc[q * 32 + k] = s;
      mx = fmaxf(mx, s);
    }
    float sum = 0.f;
    for (int k = 0; k < SEQ; ++k) {
      float e = __expf(sc[q * 32 + k] - mx);
      sc[q * 32 + k] = e;
      sum += e;
    }
    float inv = 1.0f / sum;
    for (int k = 0; k < SEQ; ++k) sc[q * 32 + k] *= inv;
  }
  __syncthreads();

  // attn_w[k] = sum_q attn[q][k] * W_attn[0][q]
  if (tid < SEQ) {
    const int k = tid;
    float a = 0.f;
    for (int q = 0; q < SEQ; ++q) a += sc[q * 32 + k] * Wattn[q];
    aw[k] = a;
    out[OUT_ATTN_OFF + (size_t)b * SEQ + k] = a;
  }
  __syncthreads();

  // context[d] = sum_k aw[k] * V[b][k][d]
  for (int d = tid; d < D_MODEL; d += 128) {
    float c = 0.f;
    for (int k = 0; k < SEQ; ++k)
      c += aw[k] * (float)Vh[(size_t)(b * SEQ + k) * D_MODEL + d];
    ctxh[(size_t)b * D_MODEL + d] = (f16)c;
  }
}

// ---------------------------------------------------------------------------
// K3: O = ctx @ W_fc^T  (+ masked-mean residual of input_Q), then LayerNorm.
// WG tile = 16 rows x all 768 cols (8 waves x 96 cols), so LN is WG-local.
// ---------------------------------------------------------------------------
__global__ __launch_bounds__(256) void out_gemm_ln(
    const f16* __restrict__ ctxh, const f16* __restrict__ Wfch,
    const float* __restrict__ inQ, const float* __restrict__ wmsk,
    float* __restrict__ out) {
  __shared__ float otile[16 * D_MODEL];   // 48 KB
  __shared__ float wsum[16];

  const int tid  = threadIdx.x;
  const int lane = tid & 31;
  const int w    = tid >> 5;
  const int half = lane >> 4;
  const int l16  = lane & 15;
  const int row0 = blockIdx.x * 16;
  const int cb   = w * 96;

  v8f acc[6];
#pragma unroll
  for (int j = 0; j < 6; ++j) acc[j] = v8f{};

  const f16* aptr = ctxh + (size_t)(row0 + l16) * D_MODEL + 8 * half;
  for (int kk = 0; kk < KSTEPS; ++kk) {
    const int k0 = kk * 32;
    HVec a;
    a.u[0] = *(const uint4*)(aptr + k0);
    a.u[1] = *(const uint4*)(aptr + k0 + 16);
#pragma unroll
    for (int j = 0; j < 6; ++j) {
      HVec bb;
      const f16* bp = Wfch + (size_t)(cb + j * 16 + l16) * D_MODEL + k0 + 8 * half;
      bb.u[0] = *(const uint4*)(bp);
      bb.u[1] = *(const uint4*)(bp + 16);
      acc[j] = __builtin_amdgcn_wmma_f32_16x16x32_f16(
          false, a.h, false, bb.h, (short)0, acc[j], false, false);
    }
  }

  // per-row word_mask sums
  if (tid < 16) {
    float s = 0.f;
    for (int ss = 0; ss < SEQ; ++ss) s += wmsk[(size_t)(row0 + tid) * SEQ + ss];
    wsum[tid] = s;
  }
  __syncthreads();

  // add masked-mean residual of input_Q, stage into LDS
#pragma unroll
  for (int j = 0; j < 6; ++j) {
#pragma unroll
    for (int v = 0; v < 8; ++v) {
      int rl = v + 8 * half;
      int r  = row0 + rl;
      int c  = cb + j * 16 + l16;
      const float* qp = inQ + (size_t)r * SEQ * D_MODEL + c;
      const float* mp = wmsk + (size_t)r * SEQ;
      float res = 0.f;
      for (int ss = 0; ss < SEQ; ++ss) res += qp[ss * D_MODEL] * mp[ss];
      otile[rl * D_MODEL + c] = acc[j][v] + res / wsum[rl];
    }
  }
  __syncthreads();

  // LayerNorm: wave w handles rows 2w and 2w+1 (wave32 shuffle reduction)
  for (int rr = 2 * w; rr < 2 * w + 2; ++rr) {
    float s = 0.f, s2 = 0.f;
    for (int c = lane; c < D_MODEL; c += 32) {
      float x = otile[rr * D_MODEL + c];
      s += x; s2 += x * x;
    }
#pragma unroll
    for (int off = 16; off > 0; off >>= 1) {
      s  += __shfl_xor(s,  off, 32);
      s2 += __shfl_xor(s2, off, 32);
    }
    float mean = s * (1.0f / D_MODEL);
    float var  = s2 * (1.0f / D_MODEL) - mean * mean;
    float rstd = rsqrtf(var + LN_EPS);
    for (int c = lane; c < D_MODEL; c += 32)
      out[(size_t)(row0 + rr) * D_MODEL + c] =
          (otile[rr * D_MODEL + c] - mean) * rstd;
  }
}

// ---------------------------------------------------------------------------
extern "C" void kernel_launch(void* const* d_in, const int* in_sizes, int n_in,
                              void* d_out, int out_size, void* d_ws, size_t ws_size,
                              hipStream_t stream) {
  const float* inQ   = (const float*)d_in[0];
  const float* inK   = (const float*)d_in[1];
  const float* inV   = (const float*)d_in[2];
  const unsigned char* amask = (const unsigned char*)d_in[3];
  const float* wmsk  = (const float*)d_in[4];
  const float* W_Q   = (const float*)d_in[5];
  const float* W_K   = (const float*)d_in[6];
  const float* W_V   = (const float*)d_in[7];
  const float* W_fc  = (const float*)d_in[8];
  const float* Wattn = (const float*)d_in[9];

  char* ws = (char*)d_ws;
  f16* whq = (f16*)(ws + WS_WQ);
  f16* whk = (f16*)(ws + WS_WK);
  f16* whv = (f16*)(ws + WS_WV);
  f16* whf = (f16*)(ws + WS_WFC);
  f16* qh  = (f16*)(ws + WS_QH);
  f16* kh  = (f16*)(ws + WS_KH);
  f16* vh  = (f16*)(ws + WS_VH);
  f16* ctx = (f16*)(ws + WS_CTX);
  float* out = (float*)d_out;

  // K0: weights f32 -> f16
  {
    int n = 4 * D_MODEL * D_MODEL;
    cvt_weights<<<(n + 255) / 256, 256, 0, stream>>>(W_Q, W_K, W_V, W_fc, whq);
  }
  // K1 x3: projections (f32 activations converted inline, f16 out)
  {
    dim3 grid(D_MODEL / 64, M_ROWS / 128);   // 12 x 576
    proj_gemm<<<grid, 256, 0, stream>>>(inQ, whq, qh);
    proj_gemm<<<grid, 256, 0, stream>>>(inK, whk, kh);
    proj_gemm<<<grid, 256, 0, stream>>>(inV, whv, vh);
  }
  // K2: per-batch attention -> attn_w (d_out tail) + context (ws)
  attention_kernel<<<BATCH, 128, 0, stream>>>(qh, kh, vh, amask, Wattn, ctx, out);
  // K3: output GEMM + residual + layernorm -> d_out head
  out_gemm_ln<<<BATCH / 16, 256, 0, stream>>>(ctx, whf, inQ, wmsk, out);
}
